// LightningAttentionLayer_42073499632265
// MI455X (gfx1250) — matrix operand order repaired and checked by reading, not compiled, not run on hardware
//
#include <hip/hip_runtime.h>

#define SEQ_L 2048
#define DMODEL 2048
#define NHEADS 32
#define HDIM 64

typedef __attribute__((ext_vector_type(16))) __bf16 v16bf;
typedef __attribute__((ext_vector_type(8)))  __bf16 v8bf;
typedef __attribute__((ext_vector_type(8)))  float  v8f;

static __device__ __forceinline__ __bf16 f2bf(float f) {
    unsigned u = __builtin_bit_cast(unsigned, f);
    unsigned r = (u + 0x7FFFu + ((u >> 16) & 1u)) >> 16;   // round-to-nearest-even
    return __builtin_bit_cast(__bf16, (unsigned short)r);
}

static __device__ __forceinline__ v16bf cat8(v8bf lo, v8bf hi) {
    return __builtin_shufflevector(lo, hi, 0,1,2,3,4,5,6,7,8,9,10,11,12,13,14,15);
}

// ------------------------------------------------- fp32 -> bf16, 8 elems/thread
__global__ void cvt_f32_to_bf16(const float* __restrict__ src,
                                __bf16* __restrict__ dst, int n8) {
    int i = blockIdx.x * blockDim.x + threadIdx.x;
    if (i >= n8) return;
    int base = i * 8;
    v8bf o;
#pragma unroll
    for (int j = 0; j < 8; ++j) o[j] = f2bf(src[base + j]);
    *(v8bf*)&dst[base] = o;
}

// ------------------------------------------------- C[MxN] = A[MxK] * B[NxK]^T
// A,B bf16 row-major; C fp32 row-major.
// Block = 4 waves; each wave computes 32(M) x 64(N) -> block tile 128 x 64.
__global__ __launch_bounds__(128)
void gemm_bf16_f32(const __bf16* __restrict__ A, const __bf16* __restrict__ B,
                   float* __restrict__ C, int M, int N, int K) {
    const int lane = threadIdx.x & 31;
    const int wave = threadIdx.x >> 5;
    const int nl = lane & 15;         // row (A) / col (B,C) within tile
    const int kb = lane >> 4;         // K sub-block selector (ISA layout)
    const int m0 = blockIdx.y * 128 + wave * 32;
    const int n0 = blockIdx.x * 64;

    v8f zero = {0.f,0.f,0.f,0.f,0.f,0.f,0.f,0.f};
    v8f acc[2][4] = {{zero, zero, zero, zero}, {zero, zero, zero, zero}};

    for (int kk = 0; kk < K; kk += 32) {
        // Two A fragments (rows m0..+15, m0+16..+31); per lane: two 16-B K runs
        v16bf afrag[2];
#pragma unroll
        for (int i = 0; i < 2; ++i) {
            const __bf16* ap = A + (size_t)(m0 + i * 16 + nl) * K + kk + kb * 8;
            afrag[i] = cat8(*(const v8bf*)ap, *(const v8bf*)(ap + 16));
        }
#pragma unroll
        for (int g = 0; g < 4; ++g) {
            // B fragment: lane nl holds col n0+g*16+nl; two 16-B K runs (same as A)
            const __bf16* bp = B + (size_t)(n0 + g * 16 + nl) * K + kk + kb * 8;
            v16bf bfrag = cat8(*(const v8bf*)bp, *(const v8bf*)(bp + 16));
#pragma unroll
            for (int i = 0; i < 2; ++i)
                acc[i][g] = __builtin_amdgcn_wmma_f32_16x16x32_bf16(
                    false, afrag[i], false, bfrag, (short)0, acc[i][g], false, false);
        }
    }
#pragma unroll
    for (int i = 0; i < 2; ++i)
#pragma unroll
        for (int g = 0; g < 4; ++g)
#pragma unroll
            for (int r = 0; r < 8; ++r)
                C[(size_t)(m0 + i * 16 + kb * 8 + r) * N + n0 + g * 16 + nl] =
                    acc[i][g][r];
}

// ----------------------------- RoPE(Q,K) -> bf16 ; V -> bf16 transposed [d][kv]
__global__ void rope_pack_kernel(const float* __restrict__ Qf,
                                 const float* __restrict__ Kf,
                                 const float* __restrict__ Vf,
                                 __bf16* __restrict__ Qb,
                                 __bf16* __restrict__ Kb,
                                 __bf16* __restrict__ Vt) {
    int idx = blockIdx.x * blockDim.x + threadIdx.x;
    if (idx >= SEQ_L * DMODEL) return;
    int row = idx / DMODEL, col = idx - row * DMODEL;
    int h = col >> 6, d = col & 63;
    int i = (d < 32) ? d : d - 32;
    // inv_freq_i = 10000^(-2i/64) = exp(-ln(1e4) * i/32)
    float inv = __expf(-9.210340371976184f * (float)i * (1.0f / 32.0f));
    float s, c;
    __sincosf((float)row * inv, &s, &c);
    int base = row * DMODEL + h * 64;
    float q1 = Qf[base + 2 * i], q2 = Qf[base + 2 * i + 1];
    float k1 = Kf[base + 2 * i], k2 = Kf[base + 2 * i + 1];
    float qv = (d < 32) ? (q1 * c - q2 * s) : (q1 * s + q2 * c);
    float kv = (d < 32) ? (k1 * c - k2 * s) : (k1 * s + k2 * c);
    Qb[idx] = f2bf(qv);
    Kb[idx] = f2bf(kv);
    Vt[(size_t)col * SEQ_L + row] = f2bf(Vf[idx]);
}

// --------------------------------------------- flash attention, 1 wave / block
// block = (q-tile of 16 rows, head). Online softmax over 64-key chunks.
__global__ __launch_bounds__(32)
void attn_kernel(const __bf16* __restrict__ Qb, const __bf16* __restrict__ Kb,
                 const __bf16* __restrict__ Vt, const int* __restrict__ mask,
                 __bf16* __restrict__ AO) {
    __shared__ __bf16 P_lds[16][64];
    const int lane = threadIdx.x & 31;
    const int nl = lane & 15, kb = lane >> 4;
    const int q0 = blockIdx.x * 16;
    const int h  = blockIdx.y;
    const float scale = 0.125f;   // 1/sqrt(64)

    // Q fragments for the two K=32 steps over head_dim=64
    v16bf qfrag[2];
#pragma unroll
    for (int s = 0; s < 2; ++s) {
        const __bf16* qp = Qb + (size_t)(q0 + nl) * DMODEL + h * 64 + s * 32 + kb * 8;
        qfrag[s] = cat8(*(const v8bf*)qp, *(const v8bf*)(qp + 16));
    }

    v8f zero = {0.f,0.f,0.f,0.f,0.f,0.f,0.f,0.f};
    v8f o[4] = {zero, zero, zero, zero};
    float mrow[8], lrow[8];
#pragma unroll
    for (int r = 0; r < 8; ++r) { mrow[r] = -INFINITY; lrow[r] = 0.f; }

    for (int kv0 = 0; kv0 < SEQ_L; kv0 += 64) {
        // ---- S = Q K^T : four 16-col tiles covering keys kv0 .. kv0+63
        v8f st[4] = {zero, zero, zero, zero};
#pragma unroll
        for (int s = 0; s < 2; ++s) {
#pragma unroll
            for (int t = 0; t < 4; ++t) {
                const __bf16* kp = Kb + (size_t)(kv0 + t * 16 + nl) * DMODEL
                                      + h * 64 + s * 32 + kb * 8;
                v16bf b = cat8(*(const v8bf*)kp, *(const v8bf*)(kp + 16));
                st[t] = __builtin_amdgcn_wmma_f32_16x16x32_bf16(
                    false, qfrag[s], false, b, (short)0, st[t], false, false);
            }
        }
        float mk[4];
#pragma unroll
        for (int t = 0; t < 4; ++t)
            mk[t] = (mask[kv0 + t * 16 + nl] == 0) ? -1.0e9f : 0.0f;

        // ---- online softmax; rows r+8*kb live in this lane's regs, row data
        //      is spread across the 16 lanes of this half -> width-16 shuffles
        float alpha[8];
#pragma unroll
        for (int r = 0; r < 8; ++r) {
            float a[4];
            float cm = -INFINITY;
#pragma unroll
            for (int t = 0; t < 4; ++t) {
                a[t] = st[t][r] * scale + mk[t];
                cm = fmaxf(cm, a[t]);
            }
#pragma unroll
            for (int off = 8; off >= 1; off >>= 1)
                cm = fmaxf(cm, __shfl_xor(cm, off, 16));
            float mn = fmaxf(mrow[r], cm);
            float al = __expf(mrow[r] - mn);
            float rs = 0.f;
#pragma unroll
            for (int t = 0; t < 4; ++t) {
                float e = __expf(a[t] - mn);
                rs += e;
                P_lds[kb * 8 + r][t * 16 + nl] = f2bf(e);   // C-layout -> LDS
            }
#pragma unroll
            for (int off = 8; off >= 1; off >>= 1)
                rs += __shfl_xor(rs, off, 16);
            lrow[r] = lrow[r] * al + rs;
            mrow[r] = mn;
            alpha[r] = al;
        }

        // ---- read P back in A-fragment layout (same-wave LDS ops in-order)
        v16bf pfrag[2];
#pragma unroll
        for (int s2 = 0; s2 < 2; ++s2)
            pfrag[s2] = cat8(*(const v8bf*)&P_lds[nl][s2 * 32 + kb * 8],
                             *(const v8bf*)&P_lds[nl][s2 * 32 + 16 + kb * 8]);

        // ---- O = O*alpha + P @ V ; Vt gives contiguous kv runs per d-column
#pragma unroll
        for (int g = 0; g < 4; ++g) {
#pragma unroll
            for (int r = 0; r < 8; ++r) o[g][r] *= alpha[r];
#pragma unroll
            for (int s2 = 0; s2 < 2; ++s2) {
                const __bf16* vp = Vt + (size_t)(h * 64 + g * 16 + nl) * SEQ_L
                                      + kv0 + s2 * 32 + kb * 8;
                v16bf vfrag = cat8(*(const v8bf*)vp, *(const v8bf*)(vp + 16));
                o[g] = __builtin_amdgcn_wmma_f32_16x16x32_bf16(
                    false, pfrag[s2], false, vfrag, (short)0, o[g], false, false);
            }
        }
    }

#pragma unroll
    for (int g = 0; g < 4; ++g)
#pragma unroll
        for (int r = 0; r < 8; ++r)
            AO[(size_t)(q0 + kb * 8 + r) * DMODEL + h * 64 + g * 16 + nl] =
                f2bf(o[g][r] / lrow[r]);
}

// -----------------------------------------------------------------------------
extern "C" void kernel_launch(void* const* d_in, const int* in_sizes, int n_in,
                              void* d_out, int out_size, void* d_ws, size_t ws_size,
                              hipStream_t stream) {
    const float* x    = (const float*)d_in[0];
    const int*   mask = (const int*)d_in[1];
    const float* Wq   = (const float*)d_in[2];
    const float* Wk   = (const float*)d_in[3];
    const float* Wv   = (const float*)d_in[4];
    const float* Wo   = (const float*)d_in[5];

    const size_t NE   = (size_t)SEQ_L * DMODEL;   // 4M elements everywhere
    const size_t SZB  = NE * sizeof(__bf16);      // 8 MB
    const size_t SZF  = NE * sizeof(float);       // 16 MB

    char* ws = (char*)d_ws;
    __bf16* xb  = (__bf16*)(ws);
    __bf16* Wqb = (__bf16*)(ws + 1 * SZB);
    __bf16* Wkb = (__bf16*)(ws + 2 * SZB);
    __bf16* Wvb = (__bf16*)(ws + 3 * SZB);
    __bf16* Wob = (__bf16*)(ws + 4 * SZB);
    float*  Qf  = (float*)(ws + 5 * SZB);
    float*  Kf  = (float*)(ws + 5 * SZB + 1 * SZF);
    float*  Vf  = (float*)(ws + 5 * SZB + 2 * SZF);
    __bf16* Qb  = (__bf16*)(ws + 5 * SZB + 3 * SZF);
    __bf16* Kb  = (__bf16*)(ws + 6 * SZB + 3 * SZF);
    __bf16* Vt  = (__bf16*)(ws + 7 * SZB + 3 * SZF);
    __bf16* AOb = (__bf16*)(ws + 8 * SZB + 3 * SZF);

    const int n8 = (int)(NE / 8);
    dim3 cblk(256), cgrd((n8 + 255) / 256);
    cvt_f32_to_bf16<<<cgrd, cblk, 0, stream>>>(x,  xb,  n8);
    cvt_f32_to_bf16<<<cgrd, cblk, 0, stream>>>(Wq, Wqb, n8);
    cvt_f32_to_bf16<<<cgrd, cblk, 0, stream>>>(Wk, Wkb, n8);
    cvt_f32_to_bf16<<<cgrd, cblk, 0, stream>>>(Wv, Wvb, n8);
    cvt_f32_to_bf16<<<cgrd, cblk, 0, stream>>>(Wo, Wob, n8);

    dim3 gblk(128), ggrd(DMODEL / 64, SEQ_L / 128);
    gemm_bf16_f32<<<ggrd, gblk, 0, stream>>>(xb, Wqb, Qf, SEQ_L, DMODEL, DMODEL);
    gemm_bf16_f32<<<ggrd, gblk, 0, stream>>>(xb, Wkb, Kf, SEQ_L, DMODEL, DMODEL);
    gemm_bf16_f32<<<ggrd, gblk, 0, stream>>>(xb, Wvb, Vf, SEQ_L, DMODEL, DMODEL);

    dim3 rgrd((unsigned)((NE + 255) / 256));
    rope_pack_kernel<<<rgrd, cblk, 0, stream>>>(Qf, Kf, Vf, Qb, Kb, Vt);

    attn_kernel<<<dim3(SEQ_L / 16, NHEADS), 32, 0, stream>>>(Qb, Kb, Vt, mask, AOb);

    gemm_bf16_f32<<<ggrd, gblk, 0, stream>>>(AOb, Wob, (float*)d_out, SEQ_L, DMODEL, DMODEL);
}
